// MOE_9010841387211
// MI455X (gfx1250) — matrix operand order repaired and checked by reading, not compiled, not run on hardware
//
#include <hip/hip_runtime.h>
#include <hip/hip_bf16.h>

#define EMBED   1024
#define NEXP    8
#define DFF     4096
#define TOK     4096          // B*S = 4*1024
#define TILE_M  32            // tokens per workgroup (2 row tiles of 16)
#define FCHUNK  128           // F columns processed per chunk

// Swizzled weight tile: 32(K) x 16(N) bf16 = 1024 B; lane L owns 32 contiguous
// bytes at tile_base + L*32 holding its 16 K-values in WMMA-B register order.
#define TILE_ELEMS 512        // bf16 elements per tile
#define W1_TILES   8192       // (DFF/16 ntiles) * (EMBED/32 ktiles) per expert
#define W2_TILES   8192       // (EMBED/16) * (DFF/32) per expert

typedef __attribute__((ext_vector_type(16))) __bf16 v16bf;
typedef __attribute__((ext_vector_type(8)))  float  v8f;

// ---------------------------------------------------------------------------
// A fragment (16x32 bf16) from LDS, this code's contiguous-K layout:
//  lane L -> row M = L&15, K-base = (L>>4)*16; element s: K=kb+s
//  (contiguous, matches swizzled B). One 16B-aligned 32B run -> 2x ds_load_b128.
// ---------------------------------------------------------------------------
__device__ __forceinline__ v16bf a_frag_lds(const __bf16* __restrict__ base,
                                            int ld, int k0, int lm, int lh) {
  const __bf16* p = base + lm * ld + k0 + (lh << 4);
  v16bf a;
#pragma unroll
  for (int i = 0; i < 4; ++i) {
    a[2 * i]         = p[2 * i];
    a[2 * i + 1]     = p[2 * i + 1];
    a[8 + 2 * i]     = p[8 + 2 * i];
    a[8 + 2 * i + 1] = p[8 + 2 * i + 1];
  }
  return a;
}

// ---------------------------------------------------------------------------
// One-time fp32 -> bf16 conversion of x.
// ---------------------------------------------------------------------------
__global__ __launch_bounds__(256)
void convert_x(const float* __restrict__ x, __bf16* __restrict__ xbf) {
  const int i = (blockIdx.x * 256 + threadIdx.x) * 4;
  const float4 v = *(const float4*)(x + i);
  __bf16* d = xbf + i;
  d[0] = (__bf16)v.x; d[1] = (__bf16)v.y; d[2] = (__bf16)v.z; d[3] = (__bf16)v.w;
}

// ---------------------------------------------------------------------------
// One-time weight swizzle: row-major fp32 [E][K][N] -> bf16 WMMA-B tiles.
// Tile index = nt*KT + kt (kt fastest => the GEMM K-loop streams contiguous
// 1KB tiles). One wave per tile; the strided gather is paid exactly once.
// ---------------------------------------------------------------------------
__global__ __launch_bounds__(256)
void swizzle_w(const float* __restrict__ w, __bf16* __restrict__ dst,
               int K, int N) {
  const int e    = blockIdx.y;
  const int KT   = K >> 5;
  const int NT   = N >> 4;
  const int tile = blockIdx.x * 8 + (threadIdx.x >> 5);   // == nt*KT + kt
  const int lane = threadIdx.x & 31;
  const int nt   = tile / KT;
  const int kt   = tile - nt * KT;
  const int n    = lane & 15;
  const int kh   = lane >> 4;

  const float* src = w + (size_t)e * K * N
                       + (size_t)(kt * 32 + kh * 16) * N + nt * 16 + n;
  v16bf b;
#pragma unroll
  for (int j = 0; j < 8; ++j) {
    b[2 * j]     = (__bf16)src[0];
    b[2 * j + 1] = (__bf16)src[N];
    src += 2 * (size_t)N;
  }
  *(v16bf*)(dst + ((size_t)e * KT * NT + tile) * TILE_ELEMS + lane * 16) = b;
}

// ---------------------------------------------------------------------------
// Router: one wave per token; renormalized top-2 softmax == softmax over the
// two largest logits. Atomic scatter into per-expert compacted lists.
// ---------------------------------------------------------------------------
__global__ __launch_bounds__(256)
void moe_router(const float* __restrict__ x,
                const float* __restrict__ rw,
                const float* __restrict__ rb,
                int*   __restrict__ cnt,
                int*   __restrict__ tlist,
                float* __restrict__ glist) {
  const int wave = threadIdx.x >> 5;
  const int lane = threadIdx.x & 31;
  const int t = blockIdx.x * 8 + wave;
  if (t >= TOK) return;

  float acc[NEXP];
#pragma unroll
  for (int e = 0; e < NEXP; ++e) acc[e] = 0.f;

  const float* xr = x + (size_t)t * EMBED;
  for (int d = lane; d < EMBED; d += 32) {
    const float xv = xr[d];
#pragma unroll
    for (int e = 0; e < NEXP; ++e) acc[e] += xv * rw[e * EMBED + d];
  }
#pragma unroll
  for (int e = 0; e < NEXP; ++e) {
#pragma unroll
    for (int m = 16; m >= 1; m >>= 1) acc[e] += __shfl_xor(acc[e], m, 32);
  }

  if (lane == 0) {
    float lg[NEXP];
#pragma unroll
    for (int e = 0; e < NEXP; ++e) lg[e] = acc[e] + rb[e];
    int i0 = 0, i1 = 1;
    if (lg[1] > lg[0]) { i0 = 1; i1 = 0; }
#pragma unroll
    for (int e = 2; e < NEXP; ++e) {
      if (lg[e] > lg[i0])      { i1 = i0; i0 = e; }
      else if (lg[e] > lg[i1]) { i1 = e; }
    }
    const float g0 = 1.f / (1.f + __expf(lg[i1] - lg[i0]));
    const float g1 = 1.f - g0;
    const int p0 = atomicAdd(&cnt[i0], 1);
    tlist[i0 * TOK + p0] = t;  glist[i0 * TOK + p0] = g0;
    const int p1 = atomicAdd(&cnt[i1], 1);
    tlist[i1 * TOK + p1] = t;  glist[i1 * TOK + p1] = g1;
  }
}

// ---------------------------------------------------------------------------
// Fused expert FFN: out[t] += gate * (relu(x[t] @ W1 + b1) @ W2 + b2)
// WG = 32 tokens of one expert, 8 waves; wave w owns out cols [w*128,+128)
// for BOTH 16-row tiles (register blocking: each weight fragment feeds 2
// WMMAs). Weights arrive as pre-swizzled bf16 fragments (2x global_load_b128,
// L2-resident). X tile gathered into LDS via async global->LDS copies.
// ---------------------------------------------------------------------------
__global__ __launch_bounds__(256)
void moe_expert_ffn(const __bf16* __restrict__ xbf,
                    const __bf16* __restrict__ w1s,
                    const float*  __restrict__ b1,
                    const __bf16* __restrict__ w2s,
                    const float*  __restrict__ b2,
                    const int*    __restrict__ cnt,
                    const int*    __restrict__ tlist,
                    const float*  __restrict__ glist,
                    float* __restrict__ out) {
  const int e = blockIdx.y;
  const int nassigned = cnt[e];
  const int tile0 = blockIdx.x * TILE_M;
  if (tile0 >= nassigned) return;

  constexpr int XLD = EMBED + 8;    // 2064B row stride (16B multiple)
  constexpr int HLD = FCHUNK + 8;
  extern __shared__ char smem[];
  __bf16* Xs = (__bf16*)smem;              // [TILE_M][XLD]  (~66 KB)
  __bf16* Hs = Xs + TILE_M * XLD;          // [TILE_M][HLD]  (~8.7 KB)

  const int tid  = threadIdx.x;
  const int wave = tid >> 5;
  const int lane = tid & 31;
  const int lm   = lane & 15;
  const int lh   = lane >> 4;

  const int*   tl = tlist + e * TOK;
  const float* gl = glist + e * TOK;
  const __bf16* W1S = w1s + (size_t)e * W1_TILES * TILE_ELEMS;
  const __bf16* W2S = w2s + (size_t)e * W2_TILES * TILE_ELEMS;

  // --- stage X tile (bf16) into LDS via async global->LDS b128 copies ---
  {
    const int row = tid >> 3;                  // 8 threads per row, 32 rows
    const int idx = tile0 + row;
    const int tok = tl[idx < nassigned ? idx : 0];
    const int c0  = (tid & 7) * 128;           // 128 bf16 = 256 B per thread
    const __bf16* src = xbf + (size_t)tok * EMBED + c0;
    const unsigned lds0 =
        (unsigned)(uintptr_t)(const void*)(Xs + row * XLD + c0);
#pragma unroll
    for (int j = 0; j < 16; ++j) {             // 16 x 16 B
      const unsigned lds_a = lds0 + j * 16;
      const unsigned long long ga =
          (unsigned long long)(uintptr_t)src + (unsigned)(j * 16);
      asm volatile("global_load_async_to_lds_b128 %0, %1, off"
                   :: "v"(lds_a), "v"(ga) : "memory");
    }
  }
  asm volatile("s_wait_asynccnt 0x0" ::: "memory");
  __syncthreads();

  const v8f vzero = {};
  v8f yacc0[8], yacc1[8];                     // 2 row tiles x 8 col tiles
#pragma unroll
  for (int i = 0; i < 8; ++i) { yacc0[i] = vzero; yacc1[i] = vzero; }

  for (int fc = 0; fc < DFF; fc += FCHUNK) {
    // ---- phase 1: H[:, fc:fc+128] = relu(X @ W1 + b1), K = 1024 ----
    {
      const int nt1 = (fc >> 4) + wave;        // this wave's H column tile
      const __bf16* w1t = W1S + ((size_t)nt1 * 32) * TILE_ELEMS + lane * 16;
      if (fc + FCHUNK < DFF)                   // prefetch next chunk's stream
        __builtin_prefetch(w1t + (size_t)8 * 32 * TILE_ELEMS + lane * 48, 0, 1);
      v8f h0 = vzero, h1 = vzero;
      for (int kk = 0; kk < EMBED; kk += 32) {
        const v16bf b  = *(const v16bf*)w1t;   // 2x global_load_b128
        w1t += TILE_ELEMS;
        const v16bf a0 = a_frag_lds(Xs, XLD, kk, lm, lh);
        const v16bf a1 = a_frag_lds(Xs + 16 * XLD, XLD, kk, lm, lh);
        h0 = __builtin_amdgcn_wmma_f32_16x16x32_bf16(
            false, a0, false, b, (short)0, h0, false, false);
        h1 = __builtin_amdgcn_wmma_f32_16x16x32_bf16(
            false, a1, false, b, (short)0, h1, false, false);
      }
      const float b1v = b1[e * DFF + fc + wave * 16 + lm];
      __bf16* hdst = Hs + (lh << 3) * HLD + wave * 16 + lm;
#pragma unroll
      for (int i = 0; i < 8; ++i) {            // C tile: VGPR i = row lh*8+i
        const float v0 = h0[i] + b1v;
        const float v1 = h1[i] + b1v;
        hdst[i * HLD]              = (__bf16)(v0 > 0.f ? v0 : 0.f);
        hdst[(16 + i) * HLD]       = (__bf16)(v1 > 0.f ? v1 : 0.f);
      }
    }
    __syncthreads();

    // ---- phase 2: Y[:, wave*128:+128] += H_chunk @ W2[fc:fc+128, :] ----
#pragma unroll
    for (int ct = 0; ct < 8; ++ct) {
      const int nt2 = wave * 8 + ct;
      const __bf16* w2t =
          W2S + ((size_t)nt2 * 128 + (fc >> 5)) * TILE_ELEMS + lane * 16;
      v8f acc0 = yacc0[ct], acc1 = yacc1[ct];
#pragma unroll
      for (int kk = 0; kk < FCHUNK; kk += 32) {
        const v16bf b  = *(const v16bf*)w2t;
        w2t += TILE_ELEMS;
        const v16bf a0 = a_frag_lds(Hs, HLD, kk, lm, lh);
        const v16bf a1 = a_frag_lds(Hs + 16 * HLD, HLD, kk, lm, lh);
        acc0 = __builtin_amdgcn_wmma_f32_16x16x32_bf16(
            false, a0, false, b, (short)0, acc0, false, false);
        acc1 = __builtin_amdgcn_wmma_f32_16x16x32_bf16(
            false, a1, false, b, (short)0, acc1, false, false);
      }
      yacc0[ct] = acc0;  yacc1[ct] = acc1;
    }
    __syncthreads();                           // Hs reused next chunk
  }

  // ---- epilogue: (+b2) * gate, atomic accumulate into out ----
  float b2v[8];
#pragma unroll
  for (int ct = 0; ct < 8; ++ct)
    b2v[ct] = b2[e * EMBED + wave * 128 + ct * 16 + lm];

#pragma unroll
  for (int rt = 0; rt < 2; ++rt) {
    const v8f* yacc = rt ? yacc1 : yacc0;
#pragma unroll
    for (int i = 0; i < 8; ++i) {
      const int row = rt * 16 + (lh << 3) + i;
      const int idx = tile0 + row;
      const bool valid = idx < nassigned;
      const int tok = tl[valid ? idx : 0];
      const float g = valid ? gl[idx] : 0.f;   // padded rows contribute 0
      float* orow = out + (size_t)tok * EMBED + wave * 128 + lm;
#pragma unroll
      for (int ct = 0; ct < 8; ++ct)
        atomicAdd(orow + ct * 16, (yacc[ct][i] + b2v[ct]) * g);
    }
  }
}

// ---------------------------------------------------------------------------
extern "C" void kernel_launch(void* const* d_in, const int* in_sizes, int n_in,
                              void* d_out, int out_size, void* d_ws, size_t ws_size,
                              hipStream_t stream) {
  const float* x  = (const float*)d_in[0];
  const float* rw = (const float*)d_in[1];
  const float* rb = (const float*)d_in[2];
  const float* w1 = (const float*)d_in[3];
  const float* b1 = (const float*)d_in[4];
  const float* w2 = (const float*)d_in[5];
  const float* b2 = (const float*)d_in[6];
  // d_in[7] is k (always 2 here; router hardcodes top-2)
  float* out = (float*)d_out;

  // Workspace layout (~136.3 MB total; all offsets 32B aligned):
  char* ws = (char*)d_ws;
  int*    cnt   = (int*)ws;                                   // 256 B
  int*    tlist = (int*)(ws + 256);                           // 128 KB
  float*  glist = (float*)(ws + 256 + 131072);                // 128 KB
  __bf16* xbf   = (__bf16*)(ws + 262400);                     // 8 MB
  __bf16* w1s   = (__bf16*)(ws + 262400 + 8388608);           // 64 MB
  __bf16* w2s   = (__bf16*)(ws + 262400 + 8388608 + 67108864);// 64 MB

  hipMemsetAsync(cnt, 0, 256, stream);
  hipMemsetAsync(out, 0, (size_t)out_size * sizeof(float), stream);

  // One-time precision/layout conversion (amortized: bf16 weights then live
  // in the 192MB L2 across all expert workgroups).
  convert_x<<<TOK * EMBED / 1024, 256, 0, stream>>>(x, xbf);
  dim3 wgrid(W1_TILES / 8, NEXP);
  swizzle_w<<<wgrid, 256, 0, stream>>>(w1, w1s, EMBED, DFF);
  swizzle_w<<<wgrid, 256, 0, stream>>>(w2, w2s, DFF, EMBED);

  moe_router<<<TOK / 8, 256, 0, stream>>>(x, rw, rb, cnt, tlist, glist);

  const size_t lds = (size_t)TILE_M * (EMBED + 8) * sizeof(__bf16)
                   + (size_t)TILE_M * (FCHUNK + 8) * sizeof(__bf16); // ~75 KB
  dim3 grid(TOK / TILE_M, NEXP);   // worst case: all tokens on one expert
  moe_expert_ffn<<<grid, 256, lds, stream>>>(xbf, w1s, b1, w2s, b2,
                                             cnt, tlist, glist, out);
}
